// CausalSelfAttention_33801392619780
// MI455X (gfx1250) — matrix operand order repaired and checked by reading, not compiled, not run on hardware
//
#include <hip/hip_runtime.h>

#define BATCH  2
#define SQ     2048
#define NH     16
#define HD     64
#define DMODEL 1024

typedef __attribute__((ext_vector_type(16))) _Float16 v16h;
typedef __attribute__((ext_vector_type(4)))  _Float16 v4h;
typedef __attribute__((ext_vector_type(8)))  float    v8f;
typedef __attribute__((ext_vector_type(4)))  float    f32x4;

union V16HU { v16h v; f32x4 q[2]; };

// A-matrix fragment (16x32 f16, row-major source, leading dim ld):
// lanes 0-15 hold row M=lane, K = 0..7 (VGPR0-3) and 16..23 (VGPR4-7)
// lanes 16-31 hold row M=lane-16, K = 8..15 and 24..31
__device__ __forceinline__ v16h load_a16(const _Float16* base, int ld, int lane) {
  const int m  = lane & 15;
  const int kh = lane >> 4;               // 0 or 1
  const _Float16* p = base + m * ld + kh * 8;
  V16HU u;
  u.q[0] = *(const f32x4*)(p);
  u.q[1] = *(const f32x4*)(p + 16);
  return u.v;
}

// B-matrix fragment (32x16 f16). B[k][n] = src[n*ld + k]  (src rows are B columns):
// lane n (0-15) holds column n, K=0..7,16..23 ; lanes 16-31 hold column n-16, K=8..15,24..31
__device__ __forceinline__ v16h load_b16(const _Float16* base, int ld, int lane) {
  const int n  = lane & 15;
  const int k0 = (lane >> 4) * 8;
  const _Float16* p = base + n * ld + k0;
  V16HU u;
  u.q[0] = *(const f32x4*)(p);
  u.q[1] = *(const f32x4*)(p + 16);
  return u.v;
}

__device__ __forceinline__ v8f wmma16(v16h a, v16h b, v8f c) {
  return __builtin_amdgcn_wmma_f32_16x16x32_f16(false, a, false, b, (short)0, c, false, false);
}

// Raw split barrier: no implicit loadcnt flush (we manage async/ds counters
// explicitly), so register prefetches survive across the barrier.
__device__ __forceinline__ void wg_barrier() {
#if __has_builtin(__builtin_amdgcn_s_barrier_signal) && __has_builtin(__builtin_amdgcn_s_barrier_wait)
  asm volatile("" ::: "memory");
  __builtin_amdgcn_s_barrier_signal(-1);
  __builtin_amdgcn_s_barrier_wait(-1);
  asm volatile("" ::: "memory");
#else
  __syncthreads();
#endif
}

// ---------------------------------------------------------------------------
// Pass 1: fp32 -> f16 conversion, 4 elements per thread (b128 in / b64 out)
// ---------------------------------------------------------------------------
__global__ __launch_bounds__(256) void cvt_kernel(
    const float* __restrict__ X,
    const float* __restrict__ Wq, const float* __restrict__ Wk, const float* __restrict__ Wv,
    _Float16* __restrict__ Xh,
    _Float16* __restrict__ Wqh, _Float16* __restrict__ Wkh, _Float16* __restrict__ Wvh) {
  const int i = blockIdx.x * 256 + threadIdx.x;       // group-of-4 index
  {
    f32x4 x = ((const f32x4*)X)[i];
    v4h h4;
    h4[0] = (_Float16)x[0]; h4[1] = (_Float16)x[1];
    h4[2] = (_Float16)x[2]; h4[3] = (_Float16)x[3];
    ((v4h*)Xh)[i] = h4;
  }
  if (i < (DMODEL * DMODEL) / 4) {
    f32x4 q = ((const f32x4*)Wq)[i];
    f32x4 k = ((const f32x4*)Wk)[i];
    f32x4 v = ((const f32x4*)Wv)[i];
    v4h hq, hk, hv;
#pragma unroll
    for (int e = 0; e < 4; ++e) {
      hq[e] = (_Float16)q[e]; hk[e] = (_Float16)k[e]; hv[e] = (_Float16)v[e];
    }
    ((v4h*)Wqh)[i] = hq; ((v4h*)Wkh)[i] = hk; ((v4h*)Wvh)[i] = hv;
  }
}

// ---------------------------------------------------------------------------
// Pass 2: QKV projection, out = X @ W^T + b.
// Workgroup (8 waves) computes 128(M) x 64(N); W tile (64 rows x 32 k, 4KB)
// staged in LDS via global_load_async_to_lds_b128; explicit 2x unroll with
// fixed buffers (no pointer swap -> no register rotation); A fragments
// software-pipelined in registers across raw split barriers.
// Q,K stored [B,H,S,DH] f16 ; V stored transposed [B,H,DH,S] f16.
// ---------------------------------------------------------------------------
__global__ __launch_bounds__(256) void qkv_gemm(
    const _Float16* __restrict__ Xh,
    const _Float16* __restrict__ Wqh, const _Float16* __restrict__ Wkh, const _Float16* __restrict__ Wvh,
    const float* __restrict__ bq, const float* __restrict__ bk, const float* __restrict__ bv,
    _Float16* __restrict__ Qh, _Float16* __restrict__ Kh, _Float16* __restrict__ Vth) {
  __shared__ __align__(16) _Float16 lb[2][64 * 32];   // double-buffered W tile

  const int tid   = threadIdx.x;
  const int lane  = tid & 31;
  const int wslot = tid >> 5;                         // 0..7

  const int mat = blockIdx.x / (32 * 16);             // 0=Q 1=K 2=V
  const int rem = blockIdx.x % (32 * 16);
  const int mt    = rem >> 4;                         // 0..31 (128 rows each)
  const int ntile = rem & 15;                         // 0..15 (64 cols each)

  const _Float16* Wh  = (mat == 0) ? Wqh : (mat == 1) ? Wkh : Wvh;
  const float*   bias = (mat == 0) ? bq  : (mat == 1) ? bk  : bv;

  // Visible init of both buffers (LDS provably written -> no undef folding).
  {
    f32x4 z = {0.f, 0.f, 0.f, 0.f};
    ((f32x4*)&lb[0][0])[tid]       = z;
    ((f32x4*)&lb[0][0])[tid + 256] = z;
  }
  __syncthreads();                                    // DS stores done before DMA writes

  // async staging: thread -> (row 0..63, 16B chunk 0..3) of the 64x32 tile
  const int srow   = tid >> 2;                        // 0..63
  const int schunk = (tid & 3) * 8;                   // half-element offset
  const _Float16* wsrc = Wh + (size_t)(ntile * 64 + srow) * DMODEL + schunk;

#define STAGE_TO(dstbase, k0)                                                  \
  do {                                                                         \
    const unsigned _dst = (unsigned)(size_t)((dstbase) + srow * 32 + schunk);  \
    asm volatile("global_load_async_to_lds_b128 %0, %1, off"                   \
                 :: "v"(_dst), "v"(wsrc + (k0)) : "memory");                   \
  } while (0)

#define COMPUTE(bufptr, areg)                                                  \
  do {                                                                         \
    v16h b0 = load_b16((bufptr) +  0 * 32, 32, lane);                          \
    v16h b1 = load_b16((bufptr) + 16 * 32, 32, lane);                          \
    v16h b2 = load_b16((bufptr) + 32 * 32, 32, lane);                          \
    v16h b3 = load_b16((bufptr) + 48 * 32, 32, lane);                          \
    acc[0] = wmma16((areg), b0, acc[0]);                                       \
    acc[1] = wmma16((areg), b1, acc[1]);                                       \
    acc[2] = wmma16((areg), b2, acc[2]);                                       \
    acc[3] = wmma16((areg), b3, acc[3]);                                       \
  } while (0)

  const _Float16* buf0 = &lb[0][0];
  const _Float16* buf1 = &lb[1][0];
  STAGE_TO(buf0, 0);                                  // prologue: k-tile 0

  const int mbase = mt * 128 + wslot * 16;
  const _Float16* Arow = Xh + (size_t)mbase * DMODEL;
  v16h a0 = load_a16(Arow, DMODEL, lane);             // prologue: A(0)
  v16h a1 = a0;
  v8f acc[4] = {};

  for (int it = 0; it < DMODEL / 32; it += 2) {
    // -- even step: compute tile it (buf0); prefetch tile it+1 (buf1) --
    STAGE_TO(buf1, (it + 1) * 32);
    a1 = load_a16(Arow + (it + 1) * 32, DMODEL, lane);
    asm volatile("s_wait_asynccnt 0x1" ::: "memory"); // buf0 DMA complete
    wg_barrier();                                     // publish buf0
    COMPUTE(buf0, a0);
    asm volatile("s_wait_dscnt 0x0" ::: "memory");    // DS reads of buf0 done
    wg_barrier();                                     // allow buf0 overwrite

    // -- odd step: compute tile it+1 (buf1); prefetch tile it+2 (buf0) --
    if (it + 2 < DMODEL / 32) {
      STAGE_TO(buf0, (it + 2) * 32);
      a0 = load_a16(Arow + (it + 2) * 32, DMODEL, lane);
      asm volatile("s_wait_asynccnt 0x1" ::: "memory");
    } else {
      asm volatile("s_wait_asynccnt 0x0" ::: "memory");
    }
    wg_barrier();                                     // publish buf1
    COMPUTE(buf1, a1);
    asm volatile("s_wait_dscnt 0x0" ::: "memory");
    wg_barrier();                                     // allow buf1 overwrite
  }
#undef COMPUTE
#undef STAGE_TO

  const int nbase  = ntile * 64 + (lane & 15);
  const int rowoff = (lane >> 4) << 3;
#pragma unroll
  for (int t = 0; t < 4; ++t) {
    const int n  = nbase + t * 16;
    const int h  = n >> 6;
    const int dd = n & 63;
    const float bv_ = bias[n];
#pragma unroll
    for (int j = 0; j < 8; ++j) {
      const int row = mbase + rowoff + j;
      const int b   = row >> 11;          // /2048
      const int s   = row & 2047;
      const float val = acc[t][j] + bv_;
      if (mat == 2)
        Vth[(((size_t)b * NH + h) * HD + dd) * SQ + s] = (_Float16)val;
      else if (mat == 0)
        Qh[(((size_t)b * NH + h) * SQ + s) * HD + dd] = (_Float16)val;
      else
        Kh[(((size_t)b * NH + h) * SQ + s) * HD + dd] = (_Float16)val;
    }
  }
}

// ---------------------------------------------------------------------------
// Pass 3: causal flash attention.  One wave per (b,h, 16-row q tile).
// K fragments issued first, V fragments second: the in-order loadcnt lets the
// score WMMAs start while V is still in flight; V latency hides under softmax.
// ---------------------------------------------------------------------------
__global__ __launch_bounds__(256) void attn_kernel(
    const _Float16* __restrict__ Qh, const _Float16* __restrict__ Kh,
    const _Float16* __restrict__ Vth, const float* __restrict__ amask,
    float* __restrict__ out) {
  __shared__ __align__(16) _Float16 lp[8][16 * 32];   // per-wave P staging tile

  const int lane  = threadIdx.x & 31;
  const int wslot = threadIdx.x >> 5;
  const int wid   = blockIdx.x * 8 + wslot;           // 0 .. B*H*(S/16)-1
  const int qt = wid & 127;                           // q tile index
  const int bh = wid >> 7;                            // 0..31
  const int b  = bh >> 4;
  const int h  = bh & 15;

  const _Float16* Qb = Qh  + (size_t)bh * SQ * HD;    // [S,DH]
  const _Float16* Kb = Kh  + (size_t)bh * SQ * HD;    // [S,DH]
  const _Float16* Vb = Vth + (size_t)bh * HD * SQ;    // [DH,S]  (V^T)

  const int q0 = qt * 16;
  const int lanecol = lane & 15;
  const int rowoff  = (lane >> 4) << 3;

  const v16h aq0 = load_a16(Qb + (size_t)q0 * HD +  0, HD, lane);
  const v16h aq1 = load_a16(Qb + (size_t)q0 * HD + 32, HD, lane);

  float mrow[8], lrow[8];
  v8f acc[4] = {};
#pragma unroll
  for (int j = 0; j < 8; ++j) { mrow[j] = -1e30f; lrow[j] = 0.f; }

  const int nkb = (q0 + 15) / 32 + 1;                 // causal: blocks 0..nkb-1
  for (int kb = 0; kb < nkb; ++kb) {
    const int c0 = kb * 32 + lanecol;

    // K fragments first (feed score WMMAs) ...
    v16h kf0 = load_b16(Kb + (size_t)(kb * 32     ) * HD,      HD, lane);
    v16h kf1 = load_b16(Kb + (size_t)(kb * 32     ) * HD + 32, HD, lane);
    v16h kf2 = load_b16(Kb + (size_t)(kb * 32 + 16) * HD,      HD, lane);
    v16h kf3 = load_b16(Kb + (size_t)(kb * 32 + 16) * HD + 32, HD, lane);
    // ... then V fragments (consumed only after softmax)
    v16h vbf0 = load_b16(Vb + (size_t)( 0) * SQ + kb * 32, SQ, lane);
    v16h vbf1 = load_b16(Vb + (size_t)(16) * SQ + kb * 32, SQ, lane);
    v16h vbf2 = load_b16(Vb + (size_t)(32) * SQ + kb * 32, SQ, lane);
    v16h vbf3 = load_b16(Vb + (size_t)(48) * SQ + kb * 32, SQ, lane);

    // ---- scores (16 x 32): two 16-wide N tiles, K-dim = 64 (2 WMMA each) ----
    v8f s0 = {}, s1 = {};
    s0 = wmma16(aq0, kf0, s0);
    s0 = wmma16(aq1, kf1, s0);
    s1 = wmma16(aq0, kf2, s1);
    s1 = wmma16(aq1, kf3, s1);

    const float am0 = amask[(size_t)b * SQ + c0];
    const float am1 = amask[(size_t)b * SQ + c0 + 16];

    float p0[8], p1[8], bmax[8];
#pragma unroll
    for (int j = 0; j < 8; ++j) {
      const int row = q0 + rowoff + j;
      float x0 = s0[j] * 0.125f + am0 + ((c0      > row) ? -10000.f : 0.f);
      float x1 = s1[j] * 0.125f + am1 + ((c0 + 16 > row) ? -10000.f : 0.f);
      p0[j] = x0; p1[j] = x1;
      bmax[j] = fmaxf(x0, x1);
    }
#pragma unroll
    for (int j = 0; j < 8; ++j) {
#pragma unroll
      for (int off = 1; off < 16; off <<= 1)
        bmax[j] = fmaxf(bmax[j], __shfl_xor(bmax[j], off, 16));
    }

    float corr[8];
#pragma unroll
    for (int j = 0; j < 8; ++j) {
      const float mn = fmaxf(mrow[j], bmax[j]);
      corr[j] = __expf(mrow[j] - mn);
      mrow[j] = mn;
      p0[j] = __expf(p0[j] - mn);
      p1[j] = __expf(p1[j] - mn);
      float ls = p0[j] + p1[j];
#pragma unroll
      for (int off = 1; off < 16; off <<= 1)
        ls += __shfl_xor(ls, off, 16);
      lrow[j] = lrow[j] * corr[j] + ls;
    }
#pragma unroll
    for (int t = 0; t < 4; ++t)
#pragma unroll
      for (int j = 0; j < 8; ++j)
        acc[t][j] *= corr[j];

    // ---- P (C-layout) -> LDS -> A-fragment (f16) ----
    _Float16* Lp = lp[wslot];
#pragma unroll
    for (int j = 0; j < 8; ++j) {
      const int r = rowoff + j;
      Lp[r * 32 + lanecol]      = (_Float16)p0[j];
      Lp[r * 32 + lanecol + 16] = (_Float16)p1[j];
    }
    asm volatile("s_wait_dscnt 0x0" ::: "memory");   // single-wave lockstep
    const v16h pa = load_a16(Lp, 32, lane);

    // ---- ctx += P @ V ----
    acc[0] = wmma16(pa, vbf0, acc[0]);
    acc[1] = wmma16(pa, vbf1, acc[1]);
    acc[2] = wmma16(pa, vbf2, acc[2]);
    acc[3] = wmma16(pa, vbf3, acc[3]);
  }

#pragma unroll
  for (int t = 0; t < 4; ++t) {
#pragma unroll
    for (int j = 0; j < 8; ++j) {
      const int row = q0 + rowoff + j;
      out[((size_t)b * SQ + row) * DMODEL + h * HD + t * 16 + lanecol] = acc[t][j] / lrow[j];
    }
  }
}

// ---------------------------------------------------------------------------
extern "C" void kernel_launch(void* const* d_in, const int* in_sizes, int n_in,
                              void* d_out, int out_size, void* d_ws, size_t ws_size,
                              hipStream_t stream) {
  const float* X     = (const float*)d_in[0];
  const float* amask = (const float*)d_in[1];
  const float* Wq    = (const float*)d_in[2];
  const float* bq    = (const float*)d_in[3];
  const float* Wk    = (const float*)d_in[4];
  const float* bk    = (const float*)d_in[5];
  const float* Wv    = (const float*)d_in[6];
  const float* bv    = (const float*)d_in[7];
  float* out = (float*)d_out;

  const size_t NX = (size_t)BATCH * SQ * DMODEL;   // 4M elements
  const size_t NW = (size_t)DMODEL * DMODEL;       // 1M elements

  _Float16* ws  = (_Float16*)d_ws;                 // needs ~38 MB of d_ws
  _Float16* Xh  = ws;
  _Float16* Wqh = Xh  + NX;
  _Float16* Wkh = Wqh + NW;
  _Float16* Wvh = Wkh + NW;
  _Float16* Qh  = Wvh + NW;
  _Float16* Kh  = Qh  + NX;
  _Float16* Vth = Kh  + NX;

  // Pass 1: convert to f16 (4 elems/thread)
  cvt_kernel<<<(int)(NX / 4 / 256), 256, 0, stream>>>(X, Wq, Wk, Wv, Xh, Wqh, Wkh, Wvh);

  // Pass 2: QKV GEMMs: 3 mats x 32 Mtiles(128) x 16 Ntiles(64) blocks
  qkv_gemm<<<3 * 32 * 16, 256, 0, stream>>>(Xh, Wqh, Wkh, Wvh, bq, bk, bv, Qh, Kh, Vth);

  // Pass 3: flash attention (B*H*(S/16) = 4096 waves, 8 waves/block)
  attn_kernel<<<(BATCH * NH * (SQ / 16)) / 8, 256, 0, stream>>>(Qh, Kh, Vth, amask, out);
}
